// TBSTANet_a_89893665505911
// MI455X (gfx1250) — matrix operand order repaired and checked by reading, not compiled, not run on hardware
//
#include <hip/hip_runtime.h>

// ---------------------------------------------------------------------------
// TBSTANet forward for MI455X (gfx1250, wave32, WMMA bf16 16x16x32).
// N=8, P=36, C=64, T=32, V=25, S=3, Ci=16.  Columns per (n,p) = T*V = 800.
// - All channel matmuls on v_wmma_f32_16x16x32_bf16 (f32 accumulate).
// - Attention-apply fused into the conv B-operand producer (LDS tile), so the
//   (N,P,192,T,V) merged tensor never touches HBM.
// - BN sum/sumsq fused into the conv epilogue (ds_add_f32 + global atomics).
// - Each wave owns two 16x16 output tiles: A fragment reused for 2 WMMAs.
// - Bulk global->LDS copies (weights, q/k slabs) via the Tensor Data Mover
//   (tensor_load_to_lds + s_wait_tensorcnt), issued once per workgroup.
// ---------------------------------------------------------------------------

#define Nn 8
#define Pp 36
#define Cc 64
#define Tt 32
#define Vv 25
#define Ss 3
#define CI 16
#define COLS 800                 // T*V
#define MNP (Nn*Pp)              // 288
#define TOT ((long)MNP*Cc*COLS)  // 14,745,600

typedef __attribute__((ext_vector_type(16))) __bf16 bf16x16;
typedef __attribute__((ext_vector_type(8)))  float  f32x8;
typedef __attribute__((ext_vector_type(4))) unsigned uint4v;
typedef __attribute__((ext_vector_type(8))) int      int8v;
typedef __attribute__((ext_vector_type(4))) int      int4v;

#if __has_builtin(__builtin_amdgcn_tensor_load_to_lds) && \
    __has_builtin(__builtin_amdgcn_s_wait_tensorcnt)
#define HAVE_TDM 1
#else
#define HAVE_TDM 0
#endif

__device__ __forceinline__ unsigned short f32_to_bf16(float f) {
  unsigned x = __float_as_uint(f);
  unsigned r = x + 0x7fffu + ((x >> 16) & 1u);   // round-to-nearest-even
  return (unsigned short)(r >> 16);
}
__device__ __forceinline__ float bf16_to_f32(unsigned short h) {
  return __uint_as_float(((unsigned)h) << 16);
}

#if HAVE_TDM
// DMA a contiguous run of `numel` bf16 elements (numel < 65536) from global
// memory into LDS via the Tensor Data Mover. D# per ISA ch.8: group0 holds
// {count=1, lds_addr, global_addr[56:0], type=2}; group1 describes a 1-row
// 2D tile: data_size=2B, tile_dim0=numel, tile_dim1=1, dim0 stride = numel.
// Generic LDS pointers carry the LDS byte offset in their low 32 bits
// (ISA 10.2).
__device__ __forceinline__ void tdm_load_bf16(const void* gsrc, void* ldst, int numel)
{
  unsigned lds = (unsigned)(unsigned long long)ldst;
  unsigned long long ga = (unsigned long long)gsrc;
  unsigned un = (unsigned)numel;
  uint4v g0;
  g0[0] = 1u;                                              // count=1 (user D#)
  g0[1] = lds;                                             // lds_addr
  g0[2] = (unsigned)(ga & 0xffffffffu);                    // global_addr[31:0]
  g0[3] = (unsigned)((ga >> 32) & 0x01ffffffu) | 0x80000000u; // ga[56:32] | type=2
  int8v g1;
  g1[0] = 0x00010000;                                      // data_size=1 (2 bytes)
  g1[1] = (int)((un & 0xffffu) << 16);                     // tensor_dim0 lo16
  g1[2] = (int)(((un >> 16) & 0xffffu) | (1u << 16));      // tensor_dim0 hi, tensor_dim1=1
  g1[3] = (int)((un & 0xffffu) << 16);                     // tile_dim0 = numel
  g1[4] = 1;                                               // tile_dim1=1, tile_dim2=0
  g1[5] = (int)un;                                         // tensor_dim0_stride lo32
  g1[6] = 0;
  g1[7] = 0;
  int4v z4 = {0, 0, 0, 0};
#if defined(__clang_major__) && (__clang_major__ >= 23)
  int8v z8 = {0, 0, 0, 0, 0, 0, 0, 0};
  __builtin_amdgcn_tensor_load_to_lds(g0, g1, z4, z4, z8, 0);
#else
  __builtin_amdgcn_tensor_load_to_lds(g0, g1, z4, z4, 0);
#endif
}
#endif

// ---------------------------------------------------------------------------
// init: convert weights to bf16 in ws, zero BN stats (must re-run each call).
// ---------------------------------------------------------------------------
__global__ __launch_bounds__(256)
void init_kernel(const float* __restrict__ W_in, const float* __restrict__ W_out,
                 const float* __restrict__ W_ff,
                 unsigned short* __restrict__ wi, unsigned short* __restrict__ wo,
                 unsigned short* __restrict__ wf, float* __restrict__ stats)
{
  int i = blockIdx.x * 256 + threadIdx.x;
  if (i < 96 * 64)  wi[i] = f32_to_bf16(W_in[i]);
  if (i < 64 * 192) wo[i] = f32_to_bf16(W_out[i]);
  if (i < 64 * 64)  wf[i] = f32_to_bf16(W_ff[i]);
  if (i < 6 * 192)  stats[i] = 0.0f;
}

// ---------------------------------------------------------------------------
// att1: positional embedding collapses over (p,t): att1[s,u,v] (3x25x25).
// ---------------------------------------------------------------------------
__global__ __launch_bounds__(256)
void att1_kernel(const float* __restrict__ W_in, const float* __restrict__ b_in,
                 const float* __restrict__ alphas, const float* __restrict__ att0,
                 float* __restrict__ att1out)
{
  __shared__ float pe[64 * 25];
  __shared__ float qk[96 * 25];
  int tid = threadIdx.x;
  for (int i = tid; i < 64 * 25; i += 256) {
    int c = i / 25, v = i % 25;
    float div = expf((float)(c & ~1) * (-9.210340371976184f / 64.0f)); // 2i*(-ln(1e4)/C)
    float ang = (float)v * div;
    pe[i] = (c & 1) ? cosf(ang) : sinf(ang);
  }
  __syncthreads();
  for (int i = tid; i < 96 * 25; i += 256) {
    int o = i / 25, v = i % 25;
    float a = b_in[o];
    for (int c = 0; c < 64; ++c) a += W_in[o * 64 + c] * pe[c * 25 + v];
    qk[i] = a;
  }
  __syncthreads();
  for (int i = tid; i < 3 * 625; i += 256) {
    int s = i / 625, r = i % 625, u = r / 25, v = r % 25;
    float a = 0.0f;
    for (int ci = 0; ci < 16; ++ci)
      a += qk[(s * 16 + ci) * 25 + u] * qk[(48 + s * 16 + ci) * 25 + v];
    // scale = 1/(Ci*T); T cancels (q,k const over t) -> dot/Ci
    att1out[i] = tanhf(a * (1.0f / 16.0f)) * alphas[s] + att0[i];
  }
}

// ---------------------------------------------------------------------------
// Fused (attention-apply +) 1x1-conv using WMMA bf16, with fused BN stats.
//   MODE 0: B[d,col] = X[n,p,d,col]                         (ff / qk conv)
//   MODE 1: B[(s,c),col] = sum_u X[n,p,c,t,u]*att1[s,u,v]   (stage1 out_proj)
//   MODE 2: same, att2[n,s,p,u,v]                           (stage2 out_proj)
//   MODE 3: B[(s,c),col] = sum_q X[n,q,c,col]*att3[n,s,p,q] (stage3 out_proj)
// Block covers 32 columns; each wave: one 16-row o-tile x two 16-col tiles.
// Weights DMA'd into LDS by wave 0 via TDM.
// ---------------------------------------------------------------------------
template <int O_CH, int D_CH, int MODE, bool OUT_BF16, bool DO_STATS>
__global__ __launch_bounds__(O_CH * 2)
void conv_wmma_kernel(const float* __restrict__ X, const unsigned short* __restrict__ Wbf,
                      const float* __restrict__ bias, const float* __restrict__ att,
                      float* __restrict__ OutF, unsigned short* __restrict__ OutB,
                      float* __restrict__ statsOut)
{
  constexpr int NW = O_CH / 16;
  constexpr int NTH = NW * 32;
  constexpr int ATT_SZ = (MODE == 1 || MODE == 2) ? 1875 : (MODE == 3 ? 108 : 1);

  const int n = blockIdx.z, p = blockIdx.y;
  const int col0 = blockIdx.x * 32;
  const int tid = threadIdx.x;
  const int lane = tid & 31, wave = tid >> 5;
  const int half = lane >> 4, r = lane & 15;
  const long base_np = (long)n * Pp + p;

  __shared__ unsigned short sW[O_CH * D_CH];       // weights, row-major (o, d)
  __shared__ unsigned short sB[32 * 32];           // [col(32)][k(32)] bf16 tile
  __shared__ float sAtt[ATT_SZ];
  __shared__ float sSum[DO_STATS ? O_CH : 1];
  __shared__ float sSq [DO_STATS ? O_CH : 1];

#if HAVE_TDM
  if (wave == 0) {                                 // one DMA per workgroup
    tdm_load_bf16(Wbf, &sW[0], O_CH * D_CH);
    __builtin_amdgcn_s_wait_tensorcnt(0);
  }
#else
  for (int i = tid; i < O_CH * D_CH; i += NTH) sW[i] = Wbf[i];
#endif
  if (DO_STATS)
    for (int i = tid; i < O_CH; i += NTH) { sSum[i] = 0.0f; sSq[i] = 0.0f; }
  if (MODE == 1) {
    for (int i = tid; i < 1875; i += NTH) sAtt[i] = att[i];
  } else if (MODE == 2) {
    for (int i = tid; i < 1875; i += NTH) {
      int s = i / 625, rr = i % 625;
      sAtt[i] = att[(((long)n * Ss + s) * Pp + p) * 625 + rr];
    }
  } else if (MODE == 3) {
    for (int i = tid; i < 108; i += NTH) {
      int s = i / 36, q = i % 36;
      sAtt[i] = att[(((long)n * Ss + s) * Pp + p) * 36 + q];
    }
  }
  __syncthreads();

  f32x8 acc0 = {0.f, 0.f, 0.f, 0.f, 0.f, 0.f, 0.f, 0.f};
  f32x8 acc1 = {0.f, 0.f, 0.f, 0.f, 0.f, 0.f, 0.f, 0.f};

  for (int d0 = 0; d0 < D_CH; d0 += 32) {
    // ---- produce B tile (32 d-rows x 32 cols) into LDS as bf16 -------------
    if (MODE == 0) {
      // float4 global loads, one d-row per lane + prefetch of next K-tile
      for (int i = tid; i < 256; i += NTH) {
        int k = i & 31, cq = i >> 5;               // cq = 0..7, 4 cols each
        int col = col0 + cq * 4;
        const float* src = X + (base_np * D_CH + d0 + k) * COLS + col;
        float4 v4 = *(const float4*)src;
        if (d0 + 32 < D_CH) __builtin_prefetch(src + 32 * COLS, 0, 3);
        sB[(cq * 4 + 0) * 32 + k] = f32_to_bf16(v4.x);
        sB[(cq * 4 + 1) * 32 + k] = f32_to_bf16(v4.y);
        sB[(cq * 4 + 2) * 32 + k] = f32_to_bf16(v4.z);
        sB[(cq * 4 + 3) * 32 + k] = f32_to_bf16(v4.w);
      }
    } else {
      for (int i = tid; i < 1024; i += NTH) {
        int k = i & 31, colI = i >> 5;
        int d = d0 + k;
        int col = col0 + colI;
        float val;
        if (MODE == 1 || MODE == 2) {
          int s = d >> 6, c = d & 63;
          int t = col / 25, vv = col % 25;
          const float* xrow = X + (base_np * Cc + c) * COLS + t * 25;
          const float* arow = sAtt + s * 625 + vv;
          float a = 0.0f;
#pragma unroll
          for (int u = 0; u < 25; ++u) a += xrow[u] * arow[u * 25];
          val = a;
        } else {  // MODE 3
          int s = d >> 6, c = d & 63;
          float a = 0.0f;
#pragma unroll 4
          for (int q = 0; q < 36; ++q)
            a += X[(((long)n * Pp + q) * Cc + c) * COLS + col] * sAtt[s * 36 + q];
          val = a;
        }
        sB[colI * 32 + k] = f32_to_bf16(val);
      }
    }
    __syncthreads();

    // ---- fragments per ISA VGPR layouts; A reused for two B tiles ----------
    union { bf16x16 v; unsigned u[8]; } afrag, bfrag0, bfrag1;
    const unsigned* sWu = (const unsigned*)sW;
    const unsigned* sBu = (const unsigned*)sB;
    const int orow = wave * 16 + r;
#pragma unroll
    for (int j = 0; j < 8; ++j) {
      // A 16x32: VGPR j holds K pair {kb,kb+1}; lanes 16-31 offset K by 8
      int kb = (j < 4) ? (2 * j + 8 * half) : (16 + 2 * (j - 4) + 8 * half);
      afrag.u[j] = sWu[(orow * D_CH + d0 + kb) >> 1];
      // B 32x16: lane = column (0-15), same K pairs as A (kb)
      bfrag0.u[j] = sBu[r * 16 + (kb >> 1)];
      bfrag1.u[j] = sBu[(16 + r) * 16 + (kb >> 1)];
    }
    acc0 = __builtin_amdgcn_wmma_f32_16x16x32_bf16(false, afrag.v, false, bfrag0.v,
                                                   (short)0, acc0, false, false);
    acc1 = __builtin_amdgcn_wmma_f32_16x16x32_bf16(false, afrag.v, false, bfrag1.v,
                                                   (short)0, acc1, false, false);
    __syncthreads();
  }

  // ---- epilogue: C layout = VGPR j -> row (j + 8*half), col = lane&15 ------
  const int colA = col0 + r;
  const int colB = col0 + 16 + r;
#pragma unroll
  for (int j = 0; j < 8; ++j) {
    int o = wave * 16 + j + 8 * half;
    float b = bias[o];
    float v0 = acc0[j] + b;
    float v1 = acc1[j] + b;
    long idx = (base_np * O_CH + o) * COLS;
    if (OUT_BF16) {
      OutB[idx + colA] = f32_to_bf16(v0);
      OutB[idx + colB] = f32_to_bf16(v1);
    } else {
      OutF[idx + colA] = v0;
      OutF[idx + colB] = v1;
    }
    if (DO_STATS) {
      atomicAdd(&sSum[o], v0 + v1);               // ds_add_f32
      atomicAdd(&sSq[o],  v0 * v0 + v1 * v1);
    }
  }
  if (DO_STATS) {
    __syncthreads();
    if (tid < O_CH) {
      atomicAdd(&statsOut[tid],      sSum[tid]);
      atomicAdd(&statsOut[96 + tid], sSq[tid]);
    }
  }
}

// ---------------------------------------------------------------------------
// BN normalize + residual + leaky-relu (fused, float4).
// ---------------------------------------------------------------------------
__global__ __launch_bounds__(256)
void bn_apply_kernel(const float4* __restrict__ O, const float4* __restrict__ Res,
                     const float* __restrict__ stats, const float* __restrict__ gamma,
                     const float* __restrict__ beta, float4* __restrict__ Out)
{
  long i = (long)blockIdx.x * 256 + threadIdx.x;
  if (i >= TOT / 4) return;
  int ch = (int)((i / 200) % Cc);                 // 200 float4 per channel row
  const float inv_cnt = 1.0f / ((float)MNP * (float)COLS);
  float m   = stats[ch] * inv_cnt;
  float var = stats[96 + ch] * inv_cnt - m * m;
  float sc  = __frsqrt_rn(var + 1e-5f) * gamma[ch];
  float sh  = beta[ch] - m * sc;
  float4 o = O[i], rr = Res[i], w;
  w.x = o.x * sc + sh + rr.x;  w.x = w.x >= 0.f ? w.x : 0.1f * w.x;
  w.y = o.y * sc + sh + rr.y;  w.y = w.y >= 0.f ? w.y : 0.1f * w.y;
  w.z = o.z * sc + sh + rr.z;  w.z = w.z >= 0.f ? w.z : 0.1f * w.z;
  w.w = o.w * sc + sh + rr.w;  w.w = w.w >= 0.f ? w.w : 0.1f * w.w;
  Out[i] = w;
}

// ---------------------------------------------------------------------------
// att2[n,s,p,u,v] = tanh(scale * sum_{ci,t} q*k)*alpha + att0 from bf16 q/k.
// q/k slabs (25.6 KB each, contiguous) DMA'd into LDS via TDM.
// ---------------------------------------------------------------------------
__global__ __launch_bounds__(256)
void att2_kernel(const unsigned short* __restrict__ QK, const float* __restrict__ alphas,
                 const float* __restrict__ att0, float* __restrict__ att2)
{
  const int p = blockIdx.x, s = blockIdx.y, n = blockIdx.z;
  __shared__ unsigned short sq[16 * COLS], sk[16 * COLS];
  const long base = ((long)n * Pp + p) * 96 * COLS;
#if HAVE_TDM
  if (threadIdx.x < 32) {
    tdm_load_bf16(QK + base + (s * 16) * COLS,      &sq[0], 16 * COLS);
    tdm_load_bf16(QK + base + (48 + s * 16) * COLS, &sk[0], 16 * COLS);
    __builtin_amdgcn_s_wait_tensorcnt(0);
  }
#else
  for (int i = threadIdx.x; i < 16 * COLS; i += 256) {
    sq[i] = QK[base + (s * 16) * COLS + i];
    sk[i] = QK[base + (48 + s * 16) * COLS + i];
  }
#endif
  __syncthreads();
  const float scale = 1.0f / (16.0f * 32.0f);
  for (int i = threadIdx.x; i < 625; i += 256) {
    int u = i / 25, v = i % 25;
    float a = 0.0f;
    for (int ci = 0; ci < 16; ++ci) {
      const unsigned short* qrow = sq + ci * COLS + u;
      const unsigned short* krow = sk + ci * COLS + v;
#pragma unroll 8
      for (int t = 0; t < 32; ++t)
        a += bf16_to_f32(qrow[t * 25]) * bf16_to_f32(krow[t * 25]);
    }
    att2[(((long)n * Ss + s) * Pp + p) * 625 + i] =
        tanhf(a * scale) * alphas[s] + att0[s * 625 + i];
  }
}

// ---------------------------------------------------------------------------
// att3[n,s,p,q] = tanh(scale * sum_{ci,t,v} q2[p]*k2[q]) * alpha[s].
// ---------------------------------------------------------------------------
__global__ __launch_bounds__(256)
void att3_kernel(const unsigned short* __restrict__ QK, const float* __restrict__ alphas,
                 float* __restrict__ att3)
{
  const int p = blockIdx.x, s = blockIdx.y, n = blockIdx.z;
  __shared__ unsigned short sq[16 * COLS];
  __shared__ float red[256];
  const long basep = (((long)n * Pp + p) * 96 + s * 16) * COLS;
#if HAVE_TDM
  if (threadIdx.x < 32) {
    tdm_load_bf16(QK + basep, &sq[0], 16 * COLS);
    __builtin_amdgcn_s_wait_tensorcnt(0);
  }
#else
  for (int i = threadIdx.x; i < 16 * COLS; i += 256) sq[i] = QK[basep + i];
#endif
  __syncthreads();
  const float scale = 1.0f / (16.0f * 32.0f);
  for (int q = 0; q < 36; ++q) {
    const long baseq = (((long)n * Pp + q) * 96 + 48 + s * 16) * COLS;
    float a = 0.0f;
    for (int i = threadIdx.x; i < 16 * COLS; i += 256)
      a += bf16_to_f32(sq[i]) * bf16_to_f32(QK[baseq + i]);
    red[threadIdx.x] = a;
    __syncthreads();
    for (int st = 128; st > 0; st >>= 1) {
      if (threadIdx.x < st) red[threadIdx.x] += red[threadIdx.x + st];
      __syncthreads();
    }
    if (threadIdx.x == 0)
      att3[(((long)n * Ss + s) * Pp + p) * 36 + q] = tanhf(red[0] * scale) * alphas[s];
    __syncthreads();
  }
}

// ---------------------------------------------------------------------------
extern "C" void kernel_launch(void* const* d_in, const int* in_sizes, int n_in,
                              void* d_out, int out_size, void* d_ws, size_t ws_size,
                              hipStream_t stream)
{
  (void)in_sizes; (void)n_in; (void)out_size; (void)ws_size;
  const float* x      = (const float*)d_in[0];
  const float* W_in   = (const float*)d_in[1];
  const float* b_in   = (const float*)d_in[2];
  const float* alphas = (const float*)d_in[3];
  const float* att0   = (const float*)d_in[4];
  const float* W_out  = (const float*)d_in[5];
  const float* b_out  = (const float*)d_in[6];
  const float* g_out  = (const float*)d_in[7];
  const float* be_out = (const float*)d_in[8];
  const float* W_ff   = (const float*)d_in[9];
  const float* b_ff   = (const float*)d_in[10];
  const float* g_ff   = (const float*)d_in[11];
  const float* be_ff  = (const float*)d_in[12];

  char* ws = (char*)d_ws;
  size_t off = 0;
  auto alloc = [&](size_t bytes) -> void* {
    void* p = ws + off;
    off = (off + bytes + 255) & ~(size_t)255;
    return p;
  };
  unsigned short* wi_bf = (unsigned short*)alloc(96 * 64 * 2);
  unsigned short* wo_bf = (unsigned short*)alloc(64 * 192 * 2);
  unsigned short* wf_bf = (unsigned short*)alloc(64 * 64 * 2);
  float* att1  = (float*)alloc(1875 * 4);
  float* stats = (float*)alloc(6 * 192 * 4);
  float* att2  = (float*)alloc((size_t)Nn * Ss * Pp * 625 * 4);
  float* att3  = (float*)alloc((size_t)Nn * Ss * Pp * Pp * 4);
  unsigned short* qk = (unsigned short*)alloc((size_t)MNP * 96 * COLS * 2);
  float* bufY = (float*)alloc((size_t)TOT * 4);
  float* bufZ = (float*)alloc((size_t)TOT * 4);
  float* bufO = (float*)alloc((size_t)TOT * 4);

  const dim3 cgrid(COLS / 32, Pp, Nn);           // 25 x 36 x 8
  const dim3 agrid(Pp, Ss, Nn);                  // attention kernels
  const int  bnBlocks = (int)((TOT / 4 + 255) / 256);

  init_kernel<<<48, 256, 0, stream>>>(W_in, W_out, W_ff, wi_bf, wo_bf, wf_bf, stats);
  att1_kernel<<<1, 256, 0, stream>>>(W_in, b_in, alphas, att0, att1);

  // ---- stage 1 -------------------------------------------------------------
  conv_wmma_kernel<64, 192, 1, false, true><<<cgrid, 128, 0, stream>>>(
      x, wo_bf, b_out, att1, bufO, nullptr, stats + 0 * 192);
  bn_apply_kernel<<<bnBlocks, 256, 0, stream>>>(
      (const float4*)bufO, (const float4*)x, stats + 0 * 192, g_out, be_out, (float4*)bufY);
  conv_wmma_kernel<64, 64, 0, false, true><<<cgrid, 128, 0, stream>>>(
      bufY, wf_bf, b_ff, nullptr, bufO, nullptr, stats + 1 * 192);
  bn_apply_kernel<<<bnBlocks, 256, 0, stream>>>(
      (const float4*)bufO, (const float4*)x, stats + 1 * 192, g_ff, be_ff, (float4*)bufY);

  // ---- stage 2 -------------------------------------------------------------
  conv_wmma_kernel<96, 64, 0, true, false><<<cgrid, 192, 0, stream>>>(
      bufY, wi_bf, b_in, nullptr, nullptr, qk, nullptr);
  att2_kernel<<<agrid, 256, 0, stream>>>(qk, alphas, att0, att2);
  conv_wmma_kernel<64, 192, 2, false, true><<<cgrid, 128, 0, stream>>>(
      bufY, wo_bf, b_out, att2, bufO, nullptr, stats + 2 * 192);
  bn_apply_kernel<<<bnBlocks, 256, 0, stream>>>(
      (const float4*)bufO, (const float4*)bufY, stats + 2 * 192, g_out, be_out, (float4*)bufZ);
  conv_wmma_kernel<64, 64, 0, false, true><<<cgrid, 128, 0, stream>>>(
      bufZ, wf_bf, b_ff, nullptr, bufO, nullptr, stats + 3 * 192);
  bn_apply_kernel<<<bnBlocks, 256, 0, stream>>>(
      (const float4*)bufO, (const float4*)bufY, stats + 3 * 192, g_ff, be_ff, (float4*)bufZ);

  // ---- stage 3 -------------------------------------------------------------
  att3_kernel<<<agrid, 256, 0, stream>>>(qk, alphas, att3);
  conv_wmma_kernel<64, 192, 3, false, true><<<cgrid, 128, 0, stream>>>(
      bufZ, wo_bf, b_out, att3, bufO, nullptr, stats + 4 * 192);
  bn_apply_kernel<<<bnBlocks, 256, 0, stream>>>(
      (const float4*)bufO, (const float4*)bufZ, stats + 4 * 192, g_out, be_out, (float4*)bufY); // v
  conv_wmma_kernel<64, 64, 0, false, true><<<cgrid, 128, 0, stream>>>(
      bufY, wf_bf, b_ff, nullptr, bufO, nullptr, stats + 5 * 192);
  bn_apply_kernel<<<bnBlocks, 256, 0, stream>>>(
      (const float4*)bufO, (const float4*)bufZ, stats + 5 * 192, g_ff, be_ff, (float4*)d_out);
}
